// ETNNMessagerLayer_60696477827106
// MI455X (gfx1250) — matrix operand order repaired and checked by reading, not compiled, not run on hardware
//
#include <hip/hip_runtime.h>
#include <hip/hip_bf16.h>

// ---------------------------------------------------------------------------
// ETNN message layer for gfx1250 (MI455X):
//   prep:  fold eval-BN into W1/b1 -> W1'^T bf16 [128][288] + b1'[128] in ws
//   cast:  (if ws fits) pre-cast x_send/x_rec/edge_attr to bf16 in ws
//   zero:  clear the [N,128] output accumulator (float4)
//   main:  per 16-edge wave tile: gather rows, v_wmma_f32_16x16x32_bf16 x 72
//          (f32 accumulate), SiLU + sigmoid gate, f32 atomic scatter-add
// ---------------------------------------------------------------------------

typedef __attribute__((ext_vector_type(16))) __bf16         v16bf;
typedef __attribute__((ext_vector_type(8)))  float          v8f;
typedef __attribute__((ext_vector_type(16))) unsigned short v16u;
typedef __attribute__((ext_vector_type(8)))  unsigned short v8u;
typedef __attribute__((ext_vector_type(4)))  unsigned short v4u;

#define DIN   272
#define DPAD  288   // 9 * 32
#define H     128

__device__ __forceinline__ unsigned short cvt_bf16(float f) {
  __bf16 h = (__bf16)f;                       // hardware RNE conversion
  return __builtin_bit_cast(unsigned short, h);
}

// ------------------------------ prep kernel --------------------------------
__global__ void etnn_prep_kernel(const float* __restrict__ gamma,
                                 const float* __restrict__ beta,
                                 const float* __restrict__ mean,
                                 const float* __restrict__ var,
                                 const float* __restrict__ W1,
                                 const float* __restrict__ b1,
                                 unsigned short* __restrict__ Wt,   // [H][DPAD] bf16
                                 float* __restrict__ b1p) {         // [H]
  __shared__ float ss[DIN];
  __shared__ float st[DIN];
  for (int k = threadIdx.x; k < DIN; k += blockDim.x) {
    float sv = gamma[k] * rsqrtf(var[k] + 1e-5f);
    ss[k] = sv;
    st[k] = beta[k] - mean[k] * sv;
  }
  __syncthreads();
  const int n = threadIdx.x;
  if (n < H) {
    float acc = b1[n];
    for (int k = 0; k < DIN; ++k) {
      float w = W1[k * H + n];
      Wt[n * DPAD + k] = cvt_bf16(ss[k] * w);
      acc += st[k] * w;
    }
    for (int k = DIN; k < DPAD; ++k) Wt[n * DPAD + k] = 0;   // zero K-padding
    b1p[n] = acc;
  }
}

// ------------------------------ cast kernel --------------------------------
__global__ void etnn_cast_kernel(const float* __restrict__ src,
                                 unsigned short* __restrict__ dst, int n4) {
  int i = blockIdx.x * blockDim.x + threadIdx.x;
  const int stride = gridDim.x * blockDim.x;
  for (; i < n4; i += stride) {
    float4 f = ((const float4*)src)[i];
    v4u o;
    o[0] = cvt_bf16(f.x); o[1] = cvt_bf16(f.y);
    o[2] = cvt_bf16(f.z); o[3] = cvt_bf16(f.w);
    ((v4u*)dst)[i] = o;
  }
}

// ------------------------------ zero kernel --------------------------------
__global__ void etnn_zero_kernel(float4* __restrict__ out, int n4) {
  int i = blockIdx.x * blockDim.x + threadIdx.x;
  const int stride = gridDim.x * blockDim.x;
  const float4 z = make_float4(0.f, 0.f, 0.f, 0.f);
  for (; i < n4; i += stride) out[i] = z;
}

// ------------------------------ main kernel --------------------------------
// PRECAST=1: gather bf16 rows directly (pure b128 loads, no VALU conversion)
// PRECAST=0: gather f32 rows and convert with hardware v_cvt in-register
template <int PRECAST>
__global__ __launch_bounds__(256) void etnn_msg_kernel(
    const float* __restrict__ xs_f, const float* __restrict__ xr_f,
    const float* __restrict__ ea_f,
    const unsigned short* __restrict__ xs_h, const unsigned short* __restrict__ xr_h,
    const unsigned short* __restrict__ ea_h,
    const int* __restrict__ index,
    const unsigned short* __restrict__ Wt, const float* __restrict__ b1p,
    const float* __restrict__ W2, const float* __restrict__ b2,
    float* __restrict__ out, int E) {
  // Stage folded weights in LDS (72 KB of the 320 KB WGP pool).
  __shared__ __align__(64) unsigned short sWt[H * DPAD];
  {
    const uint4* src = (const uint4*)Wt;
    uint4* dst = (uint4*)sWt;
    for (int i = threadIdx.x; i < (H * DPAD) / 8; i += 256) dst[i] = src[i];
  }
  __syncthreads();

  const int lane = threadIdx.x & 31;
  const int wid  = threadIdx.x >> 5;        // 8 waves -> 8 x 16 edges per block
  const int mrow = lane & 15;               // row (edge) / column (feature) in tile
  const int hiA  = (lane >> 4) * 8;         // 16-bit A-matrix K split: 0 / 8
  const int hiB  = (lane >> 4) * 8;         // 16-bit B-matrix K split: 0 / 8
  const long long eb = (long long)blockIdx.x * 128 + wid * 16;

  // Gather source rows for this lane's edge (clamped for the tail tile).
  long long e  = eb + mrow;
  long long ec = e < E ? e : (long long)(E - 1);
  const int is = index[ec];
  const int ir = index[(long long)E + ec];

  v8f acc[8] = {};   // 8 x (16x16 f32) accumulator tiles = all 128 features

  // K loop: 9 steps of 32 (x_send 0..127 | x_rec 128..255 | attr+pad 256..287)
#pragma unroll
  for (int kb = 0; kb < 9; ++kb) {
    const int k0 = kb * 32;
    v16bf a;
    if constexpr (PRECAST) {
      v16u au;
      if (kb < 8) {
        const unsigned short* base =
            (kb < 4) ? (xs_h + (long long)is * H + k0 + hiA)
                     : (xr_h + (long long)ir * H + (k0 - 128) + hiA);
        const v8u lo = *(const v8u*)base;
        const v8u hi = *(const v8u*)(base + 16);
#pragma unroll
        for (int i = 0; i < 8; ++i) { au[i] = lo[i]; au[8 + i] = hi[i]; }
      } else {
        const v8u lo = *(const v8u*)(ea_h + ec * 16 + hiA);
#pragma unroll
        for (int i = 0; i < 8; ++i) { au[i] = lo[i]; au[8 + i] = 0; }
      }
      a = __builtin_bit_cast(v16bf, au);
    } else {
      float4 f0, f1, f2, f3;
      if (kb < 4) {
        const float4* p = (const float4*)(xs_f + (long long)is * H + k0 + hiA);
        f0 = p[0]; f1 = p[1]; f2 = p[4]; f3 = p[5];
      } else if (kb < 8) {
        const float4* p = (const float4*)(xr_f + (long long)ir * H + (k0 - 128) + hiA);
        f0 = p[0]; f1 = p[1]; f2 = p[4]; f3 = p[5];
      } else {
        const float4* p = (const float4*)(ea_f + ec * 16 + hiA);
        f0 = p[0]; f1 = p[1];
        f2 = make_float4(0.f, 0.f, 0.f, 0.f);
        f3 = make_float4(0.f, 0.f, 0.f, 0.f);
      }
      a[0]  = (__bf16)f0.x; a[1]  = (__bf16)f0.y; a[2]  = (__bf16)f0.z; a[3]  = (__bf16)f0.w;
      a[4]  = (__bf16)f1.x; a[5]  = (__bf16)f1.y; a[6]  = (__bf16)f1.z; a[7]  = (__bf16)f1.w;
      a[8]  = (__bf16)f2.x; a[9]  = (__bf16)f2.y; a[10] = (__bf16)f2.z; a[11] = (__bf16)f2.w;
      a[12] = (__bf16)f3.x; a[13] = (__bf16)f3.y; a[14] = (__bf16)f3.z; a[15] = (__bf16)f3.w;
    }

#pragma unroll
    for (int t = 0; t < 8; ++t) {
      const v8u blo = *(const v8u*)&sWt[(t * 16 + mrow) * DPAD + k0 + hiB];
      const v8u bhi = *(const v8u*)&sWt[(t * 16 + mrow) * DPAD + k0 + 16 + hiB];
      v16u bu;
#pragma unroll
      for (int i = 0; i < 8; ++i) { bu[i] = blo[i]; bu[8 + i] = bhi[i]; }
      const v16bf bm = __builtin_bit_cast(v16bf, bu);
      acc[t] = __builtin_amdgcn_wmma_f32_16x16x32_bf16(
          /*neg_a=*/false, a, /*neg_b=*/false, bm,
          /*c_mod=*/(short)0, acc[t], /*reuse_a=*/false, /*reuse_b=*/false);
    }
  }

  // Epilogue: bias + SiLU, gate = sigmoid(m . W2 + b2), gated atomic scatter.
  float b1v[8], w2v[8];
#pragma unroll
  for (int t = 0; t < 8; ++t) {
    const int n = mrow + 16 * t;
    b1v[t] = b1p[n];
    w2v[t] = W2[n];
  }
#pragma unroll
  for (int t = 0; t < 8; ++t) {
#pragma unroll
    for (int j = 0; j < 8; ++j) {
      float x = acc[t][j] + b1v[t];
      acc[t][j] = x / (1.0f + __expf(-x));   // SiLU
    }
  }

  float gj[8];
  const float bb2 = b2[0];
#pragma unroll
  for (int j = 0; j < 8; ++j) {
    float s = 0.f;
#pragma unroll
    for (int t = 0; t < 8; ++t) s += acc[t][j] * w2v[t];
    // reduce over the 16 lanes that share this edge (stays within half-wave)
    s += __shfl_xor(s, 1, 32);
    s += __shfl_xor(s, 2, 32);
    s += __shfl_xor(s, 4, 32);
    s += __shfl_xor(s, 8, 32);
    gj[j] = 1.0f / (1.0f + __expf(-(s + bb2)));
  }

  const int mhi = (lane >> 4) * 8;   // which 8 edges this half-wave holds
#pragma unroll
  for (int j = 0; j < 8; ++j) {
    const long long ee = eb + mhi + j;
    if (ee < E) {
      const long long r = (long long)index[(long long)E + ee];
      float* orow = out + r * H + mrow;
#pragma unroll
      for (int t = 0; t < 8; ++t)
        unsafeAtomicAdd(orow + t * 16, acc[t][j] * gj[j]);
    }
  }
}

// ------------------------------ launcher -----------------------------------
extern "C" void kernel_launch(void* const* d_in, const int* in_sizes, int n_in,
                              void* d_out, int out_size, void* d_ws, size_t ws_size,
                              hipStream_t stream) {
  const float* x_send = (const float*)d_in[0];
  const float* x_rec  = (const float*)d_in[1];
  const int*   index  = (const int*)d_in[2];
  const float* eattr  = (const float*)d_in[3];
  const float* gamma  = (const float*)d_in[4];
  const float* beta   = (const float*)d_in[5];
  const float* mean   = (const float*)d_in[6];
  const float* var    = (const float*)d_in[7];
  const float* W1     = (const float*)d_in[8];
  const float* b1     = (const float*)d_in[9];
  const float* W2     = (const float*)d_in[10];
  const float* b2     = (const float*)d_in[11];
  float* out = (float*)d_out;

  const int N = in_sizes[0] / H;    // x_send is [N,128]
  const int E = in_sizes[3] / 16;   // edge_attr is [E,16]

  // Workspace layout (256B-aligned sections):
  //   [0, 73728)            W1'^T bf16 [128][288]
  //   [73728, 74240)        b1' f32 [128]
  //   [74240, ...)          optional bf16 pre-casts of x_send / x_rec / edge_attr
  unsigned short* Wt  = (unsigned short*)d_ws;
  float*          b1p = (float*)((char*)d_ws + H * DPAD * sizeof(unsigned short));
  const size_t base = (size_t)H * DPAD * 2 + 512;                 // 74240
  unsigned short* xs_h = (unsigned short*)((char*)d_ws + base);
  unsigned short* xr_h = xs_h + (size_t)N * H;
  unsigned short* ea_h = xr_h + (size_t)N * H;
  const size_t need = base + ((size_t)N * H * 2 + (size_t)E * 16) * sizeof(unsigned short);
  const bool precast = (ws_size >= need);

  etnn_prep_kernel<<<1, 288, 0, stream>>>(gamma, beta, mean, var, W1, b1, Wt, b1p);
  etnn_zero_kernel<<<1024, 256, 0, stream>>>((float4*)out, out_size / 4);

  const int grid = (E + 127) / 128;   // 128 edges per 256-thread block
  if (precast) {
    etnn_cast_kernel<<<2048, 256, 0, stream>>>(x_send, xs_h, N * H / 4);
    etnn_cast_kernel<<<2048, 256, 0, stream>>>(x_rec,  xr_h, N * H / 4);
    etnn_cast_kernel<<<2048, 256, 0, stream>>>(eattr,  ea_h, E * 16 / 4);
    etnn_msg_kernel<1><<<grid, 256, 0, stream>>>(nullptr, nullptr, nullptr,
                                                 xs_h, xr_h, ea_h, index,
                                                 Wt, b1p, W2, b2, out, E);
  } else {
    etnn_msg_kernel<0><<<grid, 256, 0, stream>>>(x_send, x_rec, eattr,
                                                 nullptr, nullptr, nullptr, index,
                                                 Wt, b1p, W2, b2, out, E);
  }
}
